// RefBasicVSRLRFeatConvAttnLargeNet_9371618640383
// MI455X (gfx1250) — matrix operand order repaired and checked by reading, not compiled, not run on hardware
//
#include <hip/hip_runtime.h>
#include <hip/hip_bf16.h>
#include <math.h>

// ---- problem constants -----------------------------------------------------
#define HH 96
#define WW 96
#define HO 94            // 96 - 3 + 1
#define NP 8836          // 94*94 patches
#define DD 2304          // 256 * 3 * 3
#define KSTEPS 72        // 2304 / 32 (K per bf16 WMMA)
#define NT_M 553         // ceil(8836/16) m-tiles
#define NTA 554          // A padded to even #tiles (MT=2 per wave)
#define NTB 556          // B padded to multiple of 2 (NT=2 per group) w/ zero tiles
#define FRAG 512         // elements per fragment: 32 lanes * 16 bf16

typedef __attribute__((ext_vector_type(16))) __bf16 v16bf;
typedef __attribute__((ext_vector_type(8)))  float  v8f;

// value of patch p, feature d (d ordered (c, dy, dx)) relative to feat base
__device__ __forceinline__ float patch_val(const float* __restrict__ base, int d) {
    int c = d / 9, r = d % 9;
    return base[(c * HH + r / 3) * WW + (r % 3)];
}

// ---- kernel 1: per-patch L2 norms (one wave per patch) ---------------------
__global__ void norms_kernel(const float* __restrict__ fin, const float* __restrict__ fref,
                             float* __restrict__ inpN, float* __restrict__ refN) {
    int wave = (blockIdx.x * blockDim.x + threadIdx.x) >> 5;
    int lane = threadIdx.x & 31;
    if (wave >= NP) return;
    int iy = wave / HO, ix = wave % HO;
    const float* bi = fin  + (iy * WW + ix);
    const float* br = fref + (iy * WW + ix);
    float si = 0.f, sr = 0.f;
    for (int d = lane; d < DD; d += 32) {
        float a = patch_val(bi, d);
        float b = patch_val(br, d);
        si += a * a;
        sr += b * b;
    }
    #pragma unroll
    for (int o = 16; o; o >>= 1) {
        si += __shfl_xor(si, o, 32);
        sr += __shfl_xor(sr, o, 32);
    }
    if (lane == 0) {
        inpN[wave] = sqrtf(si) + 1e-5f;
        refN[wave] = sqrtf(sr) + 1e-5f;
    }
}

// ---- kernel 2: pack input patches into bf16 A-fragment layout --------------
// A 16x32 bf16 packing used here (contiguous k, same as B): lanes 0-15 -> M=lane, K = e;
//                      lanes 16-31 -> M=lane-16, K = 16+e
__global__ void pack_a(const float* __restrict__ fin, __bf16* __restrict__ A) {
    int t = blockIdx.x * blockDim.x + threadIdx.x;
    if (t >= NTA * KSTEPS * 32) return;
    int lane = t & 31;
    int kk   = (t >> 5) % KSTEPS;
    int tm   = t / (KSTEPS * 32);
    int m = tm * 16 + (lane & 15);
    bool valid = m < NP;
    const float* base = fin;
    if (valid) base = fin + ((m / HO) * WW + (m % HO));
    int khalf = (lane < 16) ? 0 : 16;
    v16bf v;
    #pragma unroll
    for (int e = 0; e < 16; ++e) {
        int k = kk * 32 + khalf + e;
        float f = valid ? patch_val(base, k) : 0.f;
        v[e] = (__bf16)f;
    }
    ((v16bf*)A)[t] = v;  // t == ((tm*KSTEPS + kk)*32 + lane)
}

// ---- kernel 3: pack L2-normalized ref patches into bf16 B-fragment layout --
// B 32x16 bf16 packing (contiguous k, as A): lane = N (mod 16); lanes 0-15 -> K = e, lanes 16-31 -> K = 16+e
__global__ void pack_b(const float* __restrict__ fref, const float* __restrict__ refN,
                       __bf16* __restrict__ B) {
    int t = blockIdx.x * blockDim.x + threadIdx.x;
    if (t >= NTB * KSTEPS * 32) return;
    int lane = t & 31;
    int kk   = (t >> 5) % KSTEPS;
    int tn   = t / (KSTEPS * 32);
    int n = tn * 16 + (lane & 15);
    bool valid = n < NP;
    const float* base = fref;
    float scale = 0.f;
    if (valid) {
        base  = fref + ((n / HO) * WW + (n % HO));
        scale = 1.f / refN[n];
    }
    int kbase = kk * 32 + ((lane < 16) ? 0 : 16);
    v16bf v;
    #pragma unroll
    for (int e = 0; e < 16; ++e) {
        float f = valid ? patch_val(base, kbase + e) * scale : 0.f;
        v[e] = (__bf16)f;
    }
    ((v16bf*)B)[t] = v;
}

// ---- kernel 4: fused WMMA GEMM + streaming argmax --------------------------
// One wave owns 2 m-tiles (32 rows); loops over all n in groups of 2 tiles.
// C/D f32 layout: VGPR r -> M = r (lanes 0-15) or 8+r (lanes 16-31); N = lane&15.
__global__ __launch_bounds__(256)
void gemm_argmax(const __bf16* __restrict__ Ab, const __bf16* __restrict__ Bb,
                 const float* __restrict__ inpN,
                 float* __restrict__ outIdxF, float* __restrict__ outVal,
                 int* __restrict__ idxBuf) {
    int lane = threadIdx.x & 31;
    int slot = (blockIdx.x * blockDim.x + threadIdx.x) >> 5;  // wave slot
    if (slot >= (NTA / 2)) return;                            // 277 active wave slots
    int tm0 = slot * 2;

    const v16bf* A0 = (const v16bf*)Ab + (size_t)tm0 * KSTEPS * 32 + lane;
    const v16bf* A1 = A0 + (size_t)KSTEPS * 32;

    float rv[16];
    int   ri[16];
    #pragma unroll
    for (int i = 0; i < 16; ++i) { rv[i] = -3.402823466e38f; ri[i] = 0; }

    int col = lane & 15;

    for (int g = 0; g < NTB / 2; ++g) {          // 278 n-groups of 2 tiles
        int tn0 = g * 2;
        const v16bf* B0 = (const v16bf*)Bb + (size_t)tn0 * KSTEPS * 32 + lane;
        const v16bf* B1 = B0 + (size_t)KSTEPS * 32;

        v8f acc00 = {}, acc01 = {}, acc10 = {}, acc11 = {};
        for (int kk = 0; kk < KSTEPS; ++kk) {
            v16bf a0 = A0[kk * 32];
            v16bf a1 = A1[kk * 32];
            v16bf b0 = B0[kk * 32];
            v16bf b1 = B1[kk * 32];
            acc00 = __builtin_amdgcn_wmma_f32_16x16x32_bf16(false, a0, false, b0,
                                                            (short)0, acc00, false, false);
            acc01 = __builtin_amdgcn_wmma_f32_16x16x32_bf16(false, a0, false, b1,
                                                            (short)0, acc01, false, false);
            acc10 = __builtin_amdgcn_wmma_f32_16x16x32_bf16(false, a1, false, b0,
                                                            (short)0, acc10, false, false);
            acc11 = __builtin_amdgcn_wmma_f32_16x16x32_bf16(false, a1, false, b1,
                                                            (short)0, acc11, false, false);
        }

        int n0 = tn0 * 16 + col;
        int n1 = n0 + 16;
        bool ok0 = n0 < NP, ok1 = n1 < NP;
        #pragma unroll
        for (int r = 0; r < 8; ++r) {
            if (ok0 && acc00[r] > rv[r])     { rv[r]     = acc00[r]; ri[r]     = n0; }
            if (ok1 && acc01[r] > rv[r])     { rv[r]     = acc01[r]; ri[r]     = n1; }
            if (ok0 && acc10[r] > rv[8 + r]) { rv[8 + r] = acc10[r]; ri[8 + r] = n0; }
            if (ok1 && acc11[r] > rv[8 + r]) { rv[8 + r] = acc11[r]; ri[8 + r] = n1; }
        }
    }

    // cross-lane argmax over the 16 columns of each half-wave (ties -> smaller n)
    #pragma unroll
    for (int i = 0; i < 16; ++i) {
        #pragma unroll
        for (int o = 1; o < 16; o <<= 1) {
            float ov = __shfl_xor(rv[i], o, 32);
            int   oi = __shfl_xor(ri[i], o, 32);
            if (ov > rv[i] || (ov == rv[i] && oi < ri[i])) { rv[i] = ov; ri[i] = oi; }
        }
    }

    if (col == 0) {
        int half = (lane < 16) ? 0 : 8;
        #pragma unroll
        for (int i = 0; i < 16; ++i) {
            int mt = (i < 8) ? tm0 : (tm0 + 1);
            int m  = mt * 16 + half + (i & 7);
            if (m < NP) {
                outVal[m]  = rv[i] / inpN[m];   // inpN already includes +EPS
                outIdxF[m] = (float)ri[i];
                idxBuf[m]  = ri[i];
            }
        }
    }
}

// ---- kernel 5: gather winning raw ref patch per output pixel ---------------
__global__ void gather_kernel(const float* __restrict__ fref, const int* __restrict__ idxBuf,
                              float* __restrict__ outSwap) {
    int m = blockIdx.x;
    int n = idxBuf[m];
    const float* base = fref + ((n / HO) * WW + (n % HO));
    float* o = outSwap + (size_t)m * DD;
    for (int d = threadIdx.x; d < DD; d += blockDim.x)
        o[d] = patch_val(base, d);
}

// ---- launch ----------------------------------------------------------------
extern "C" void kernel_launch(void* const* d_in, const int* in_sizes, int n_in,
                              void* d_out, int out_size, void* d_ws, size_t ws_size,
                              hipStream_t stream) {
    const float* fin  = (const float*)d_in[0];
    const float* fref = (const float*)d_in[1];
    float* out = (float*)d_out;

    // workspace: A (554*72*512 bf16) | B (556*72*512 bf16) | inpN | refN | idx
    __bf16* A = (__bf16*)d_ws;
    __bf16* B = A + (size_t)NTA * KSTEPS * FRAG;
    float* inpN = (float*)(B + (size_t)NTB * KSTEPS * FRAG);
    float* refN = inpN + NP;
    int*   idxB = (int*)(refN + NP);

    float* outIdxF = out;             // (94,94) argmax as float
    float* outVal  = out + NP;        // (94,94) normalized max corr
    float* outSwap = out + 2 * NP;    // (94,94,256,3,3) gathered patches

    norms_kernel<<<(NP * 32 + 255) / 256, 256, 0, stream>>>(fin, fref, inpN, refN);

    int ta = NTA * KSTEPS * 32;
    pack_a<<<(ta + 255) / 256, 256, 0, stream>>>(fin, A);
    int tb = NTB * KSTEPS * 32;
    pack_b<<<(tb + 255) / 256, 256, 0, stream>>>(fref, refN, B);

    // 277 wave-slots, 8 waves / block -> 35 blocks
    gemm_argmax<<<35, 256, 0, stream>>>(A, B, inpN, outIdxF, outVal, idxB);

    gather_kernel<<<NP, 256, 0, stream>>>(fref, idxB, outSwap);
}
